// HAttentionLayer_28114855919649
// MI455X (gfx1250) — matrix operand order repaired and checked by reading, not compiled, not run on hardware
//
#include <hip/hip_runtime.h>

typedef __attribute__((ext_vector_type(16))) _Float16 v16h;
typedef __attribute__((ext_vector_type(8)))  float    v8f;

#define BB 2
#define NN 2048
#define FF 16
#define HH 4
#define DA 8
#define DO 16
#define CC 2
#define CO 2
#define HIDN 4
#define NCHUNK (NN/32)
#define EPSF 1e-7f

// ---- workspace layout (float offsets) ----
#define OFF_SRCQ 0u
#define OFF_DSTQ 32768u
#define OFF_SRCV 65536u
#define OFF_DSTV 98304u
#define OFF_RMAX 131072u
#define OFF_RSUM 196608u
#define OFF_QATT 262144u      // [C][B][N][H][DA]
#define OFF_VRAW 524288u      // [C][B][4][N][DO]
#define OFF_QBUF 1048576u     // [C][B][N][H][DA]
#define OFF_SQ   1310720u     // [C][B][N][H]
#define OFF_MASK 1343488u     // [C][B][N][N]
#define OFF_HFRAG_BYTES 139591680ull  // ( (OFF_MASK+33554432)*4 ), 32B aligned

__device__ __forceinline__ float ftanh(float x) {
  x = fminf(fmaxf(x, -15.f), 15.f);
  float e = __expf(2.f * x);
  return (e - 1.f) / (e + 1.f);
}
__device__ __forceinline__ float fartanh(float x) {
  x = fminf(fmaxf(x, -1.f + 1e-6f), 1.f - 1e-6f);
  return 0.5f * __logf((1.f + x) / (1.f - x));
}
__device__ __forceinline__ float felu(float x) {
  return x > 0.f ? x : (__expf(x) - 1.f);
}

// ---------------------------------------------------------------------------
// K1: xt = logmap0(x); h_q = xt@Wq, h_v = xt@Wv per channel; src/dst scalar
// projections; pack h into f16 WMMA-B fragment order for K3.
// One wave per row. grid = 512 x 256.
// ---------------------------------------------------------------------------
__global__ void k1_proj(const float* __restrict__ x,
                        const float* __restrict__ Wq, const float* __restrict__ aqs,
                        const float* __restrict__ aqd,
                        const float* __restrict__ Wv, const float* __restrict__ avs,
                        const float* __restrict__ avd,
                        float* __restrict__ srcq, float* __restrict__ dstq,
                        float* __restrict__ srcv, float* __restrict__ dstv,
                        _Float16* __restrict__ hfrag) {
  __shared__ float xtsh[8][16];
  __shared__ float hsh[8][96];
  int wave = threadIdx.x >> 5, lane = threadIdx.x & 31;
  int row = blockIdx.x * 8 + wave;
  int b = row / NN, n = row % NN;

  float xv = (lane < FF) ? x[row * FF + lane] : 0.f;
  float s = xv * xv;
  for (int off = 16; off >= 1; off >>= 1) s += __shfl_xor(s, off);
  float nr = fmaxf(sqrtf(s), EPSF);
  float sc = fartanh(nr) / nr;
  if (lane < FF) xtsh[wave][lane] = sc * xv;
  __syncthreads();

  int chunk = n >> 5, k = n & 31;
  int khi = (k >= 16) ? 1 : 0, kel = k & 15;

  for (int c = 0; c < CC; ++c) {
    float hq = 0.f, hv0 = 0.f, hv1 = 0.f;
    for (int f = 0; f < FF; ++f) {
      float xt = xtsh[wave][f];
      hq  += xt * Wq[(c * FF + f) * (HH * DA) + lane];
      hv0 += xt * Wv[(c * FF + f) * (HH * DO) + lane];
      hv1 += xt * Wv[(c * FF + f) * (HH * DO) + lane + 32];
    }
    hsh[wave][lane]      = hq;
    hsh[wave][32 + lane] = hv0;
    hsh[wave][64 + lane] = hv1;
    __syncthreads();

    if (lane < HH) {                         // src/dst for q heads
      int h = lane; float sv = 0.f, dv = 0.f;
      for (int d = 0; d < DA; ++d) {
        float hval = hsh[wave][h * DA + d];
        sv += hval * aqs[(c * HH + h) * DA + d];
        dv += hval * aqd[(c * HH + h) * DA + d];
      }
      srcq[((c * BB + b) * HH + h) * NN + n] = sv;
      dstq[((c * BB + b) * HH + h) * NN + n] = dv;
    } else if (lane < 2 * HH) {              // src/dst for v heads
      int h = lane - HH; float sv = 0.f, dv = 0.f;
      for (int d = 0; d < DO; ++d) {
        float hval = hsh[wave][32 + h * DO + d];
        sv += hval * avs[(c * HH + h) * DO + d];
        dv += hval * avd[(c * HH + h) * DO + d];
      }
      srcv[((c * BB + b) * HH + h) * NN + n] = sv;
      dstv[((c * BB + b) * HH + h) * NN + n] = dv;
    }

    // Pack B fragments: value h[row m][head][d] -> lane_f = d + 16*(k>=16),
    // element within v16h = k%16 (VGPR j=(k%16)/2, half=k&1).
    size_t fb = ((size_t)((c * BB + b) * NCHUNK + chunk)) * 8;  // + head
    {   // q heads: 4 heads x Da=8, zero-pad d=8..15
      int hd = lane >> 3, d = lane & 7;
      size_t base = (fb + hd) * 32;
      hfrag[(base + (d + 16 * khi)) * 16 + kel]     = (_Float16)hsh[wave][lane];
      hfrag[(base + (d + 8 + 16 * khi)) * 16 + kel] = (_Float16)0.f;
    }
    for (int t2 = 0; t2 < 2; ++t2) {  // v heads: 4 heads x Do=16
      int cv = lane + 32 * t2;
      int hd = 4 + (cv >> 4), d = cv & 15;
      size_t base = (fb + hd) * 32;
      hfrag[(base + (d + 16 * khi)) * 16 + kel] = (_Float16)hsh[wave][32 + cv];
    }
    __syncthreads();
  }
}

// ---------------------------------------------------------------------------
// K2: fused 8-head online softmax stats (rowmax, rowsum). One adj pass,
// float4 streaming reads, branchless -1e9 masked logits (faithful to the
// reference even in the all-masked corner case). grid = 1024 x 256.
// ---------------------------------------------------------------------------
__global__ void k2_stats(const float* __restrict__ adj,
                         const float* __restrict__ srcq, const float* __restrict__ dstq,
                         const float* __restrict__ srcv, const float* __restrict__ dstv,
                         float* __restrict__ rmax, float* __restrict__ rsum) {
  __shared__ float dsh[8][NN];
  int rb = blockIdx.x & 255;
  int b  = (blockIdx.x >> 8) & 1;
  int c  = blockIdx.x >> 9;
  for (int i = threadIdx.x; i < 8 * NN; i += 256) {
    int hd = i >> 11, m = i & (NN - 1);
    const float* da = (hd < HH) ? dstq : dstv;
    dsh[hd][m] = da[((c * BB + b) * HH + (hd & 3)) * NN + m];
  }
  __syncthreads();
  int wave = threadIdx.x >> 5, lane = threadIdx.x & 31;
  int n = rb * 8 + wave;
  float sval[8], mx[8], sm[8];
#pragma unroll
  for (int hd = 0; hd < 8; ++hd) {
    const float* sa = (hd < HH) ? srcq : srcv;
    sval[hd] = sa[((c * BB + b) * HH + (hd & 3)) * NN + n];
    mx[hd] = -1e38f; sm[hd] = 0.f;
  }
  const float* arow = adj + ((size_t)(b * CC + c) * NN + n) * NN;
  const float4* arow4 = (const float4*)arow;
  for (int i4 = lane; i4 < NN / 4; i4 += 32) {
    float4 a4 = arow4[i4];
    float av[4] = {a4.x, a4.y, a4.z, a4.w};
#pragma unroll
    for (int j = 0; j < 4; ++j) {
      int m = i4 * 4 + j;
      bool ok = av[j] > 0.f;
#pragma unroll
      for (int hd = 0; hd < 8; ++hd) {
        float e = sval[hd] + dsh[hd][m];
        e = e > 0.f ? e : 0.2f * e;
        e = ok ? e : -1e9f;
        float nm = fmaxf(mx[hd], e);
        sm[hd] = sm[hd] * __expf(mx[hd] - nm) + __expf(e - nm);
        mx[hd] = nm;
      }
    }
  }
#pragma unroll
  for (int hd = 0; hd < 8; ++hd) {
    float m0 = mx[hd], s0 = sm[hd];
    for (int off = 16; off >= 1; off >>= 1) {
      float m1 = __shfl_xor(m0, off), s1 = __shfl_xor(s0, off);
      float nm = fmaxf(m0, m1);
      s0 = s0 * __expf(m0 - nm) + s1 * __expf(m1 - nm);
      m0 = nm;
    }
    if (lane == 0) {
      rmax[((c * BB + b) * 8 + hd) * NN + n] = m0;
      rsum[((c * BB + b) * 8 + hd) * NN + n] = s0;
    }
  }
}

// ---------------------------------------------------------------------------
// K3: attention P@V via v_wmma_f32_16x16x32_f16. One adj pass shared by all
// 8 heads (8 waves/block, one head per wave, 16-row tile, K=32 chunks).
// Double-buffered LDS stages (1 barrier/iter), pipelined B-fragment loads,
// global_prefetch of adj 8 chunks ahead. grid = C*B*128 x 256.
// ---------------------------------------------------------------------------
__global__ void k3_attn(const float* __restrict__ adj,
                        const float* __restrict__ srcq, const float* __restrict__ srcv,
                        const float* __restrict__ dstq, const float* __restrict__ dstv,
                        const float* __restrict__ rmax, const float* __restrict__ rsum,
                        const _Float16* __restrict__ hfrag,
                        float* __restrict__ qatt, float* __restrict__ vraw4) {
  __shared__ float adjsh[2][16][33];
  __shared__ float dstsh[2][8][32];
  int rb = blockIdx.x & 127;
  int b  = (blockIdx.x >> 7) & 1;
  int c  = blockIdx.x >> 8;
  int n0 = rb * 16;
  int wave = threadIdx.x >> 5, lane = threadIdx.x & 31;
  int hd = wave;
  int r = lane & 15, hi = lane >> 4;

  const float* sa = (hd < HH) ? srcq : srcv;
  float srcval = sa[((c * BB + b) * HH + (hd & 3)) * NN + n0 + r];
  float rmx = rmax[((c * BB + b) * 8 + hd) * NN + n0 + r];
  float rsm = rsum[((c * BB + b) * 8 + hd) * NN + n0 + r];
  float inv = 1.f / fmaxf(rsm, 1e-30f);

  const float* adjb = adj + ((size_t)(b * CC + c) * NN + n0) * NN;
  const v16h* fragp = (const v16h*)hfrag;
  size_t fragbase = (size_t)(c * BB + b) * NCHUNK;

  // stage loader: 16x32 adj tile as float2 per thread + per-head dst chunk
  int rowi = threadIdx.x >> 4;
  int col2 = (threadIdx.x & 15) * 2;
  int lh = threadIdx.x >> 5, lm = threadIdx.x & 31;
  const float* dla = (lh < HH) ? dstq : dstv;
  const float* dlp = dla + ((c * BB + b) * HH + (lh & 3)) * NN;

  auto load_stage = [&](int chunk, int sbuf) {
    int k0 = chunk * 32;
    const float2* srcp = (const float2*)(adjb + (size_t)rowi * NN + k0);
    float2 v2 = srcp[col2 >> 1];
    adjsh[sbuf][rowi][col2]     = v2.x;
    adjsh[sbuf][rowi][col2 + 1] = v2.y;
    dstsh[sbuf][lh][lm] = dlp[k0 + lm];
  };

  union { v8f v; float f[8]; } acc = {};

  load_stage(0, 0);
  v16h bf = fragp[((fragbase + 0) * 8 + hd) * 32 + lane];
  __syncthreads();

  for (int chunk = 0; chunk < NCHUNK; ++chunk) {
    int cur = chunk & 1;
    if (chunk + 1 < NCHUNK) load_stage(chunk + 1, cur ^ 1);
    v16h bfn = bf;
    if (chunk + 1 < NCHUNK)
      bfn = fragp[((fragbase + chunk + 1) * 8 + hd) * 32 + lane];
    if (chunk + 8 < NCHUNK)
      __builtin_prefetch(adjb + (size_t)rowi * NN + (chunk + 8) * 32, 0, 0);

    union { v16h v; _Float16 h[16]; } af;
#pragma unroll
    for (int j = 0; j < 8; ++j) {
#pragma unroll
      for (int half = 0; half < 2; ++half) {
        // A layout: lanes0-15 K in {0..15}; lanes16-31 K in {16..31}
        int kk = 16 * hi + 2 * j + half;
        float a = adjsh[cur][r][kk];
        float e = srcval + dstsh[cur][hd][kk];
        e = e > 0.f ? e : 0.2f * e;
        e = (a > 0.f) ? e : -1e9f;            // branchless masked logit
        float p = __expf(e - rmx) * inv;
        af.h[2 * j + half] = (_Float16)p;
      }
    }
    acc.v = __builtin_amdgcn_wmma_f32_16x16x32_f16(
        false, af.v, false, bf, (short)0, acc.v, false, false);
    bf = bfn;
    __syncthreads();
  }

#pragma unroll
  for (int j = 0; j < 8; ++j) {
    int orow = n0 + j + 8 * hi;     // C layout: VGPR j -> M=j (+8 for hi lanes)
    int col = lane & 15;
    if (hd < HH) {
      if (col < DA)
        qatt[(((size_t)(c * BB + b) * NN + orow) * HH + hd) * DA + col] = acc.f[j];
    } else {
      vraw4[(((size_t)(c * BB + b) * HH + (hd - 4)) * NN + orow) * DO + col] = acc.f[j];
    }
  }
}

// ---------------------------------------------------------------------------
// K4: per-row: q = expmap0(qatt) (+|q|^2), v = expmap0(mean_h), xs = logmap0(v),
// channel-mix MLP, x_out = expmap0(out*flags). grid = 16 x 256.
// ---------------------------------------------------------------------------
__global__ void k4_rows(const float* __restrict__ qattb, const float* __restrict__ vraw4,
                        const float* __restrict__ flags,
                        const float* __restrict__ mcW1, const float* __restrict__ mcb1,
                        const float* __restrict__ mcW2, const float* __restrict__ mcb2,
                        float* __restrict__ qbuf, float* __restrict__ sqbuf,
                        float* __restrict__ xout) {
  int row = blockIdx.x * 256 + threadIdx.x;
  int b = row / NN, n = row % NN;
  float xcat[CC * DO];
  for (int c = 0; c < CC; ++c) {
    for (int h = 0; h < HH; ++h) {
      size_t qi = (((size_t)(c * BB + b) * NN + n) * HH + h) * DA;
      float qr[DA]; float s2 = 0.f;
      for (int d = 0; d < DA; ++d) { qr[d] = qattb[qi + d]; s2 += qr[d] * qr[d]; }
      float nr = fmaxf(sqrtf(s2), EPSF);
      float scl = ftanh(nr) / nr;
      float sq = 0.f;
      for (int d = 0; d < DA; ++d) {
        float qq = scl * qr[d]; qbuf[qi + d] = qq; sq += qq * qq;
      }
      sqbuf[((size_t)(c * BB + b) * NN + n) * HH + h] = sq;
    }
    float vt[DO];
    for (int d = 0; d < DO; ++d) vt[d] = 0.f;
    for (int p = 0; p < HH; ++p) {
      size_t vi = (((size_t)(c * BB + b) * HH + p) * NN + n) * DO;
      for (int d = 0; d < DO; ++d) vt[d] += vraw4[vi + d];
    }
    float s2 = 0.f;
    for (int d = 0; d < DO; ++d) { vt[d] *= 0.25f; s2 += vt[d] * vt[d]; }
    float nr = fmaxf(sqrtf(s2), EPSF);
    float scl = ftanh(nr) / nr;
    float vv[DO]; float s3 = 0.f;
    for (int d = 0; d < DO; ++d) { vv[d] = scl * vt[d]; s3 += vv[d] * vv[d]; }
    float nv = fmaxf(sqrtf(s3), EPSF);
    float scl2 = fartanh(nv) / nv;
    for (int d = 0; d < DO; ++d) xcat[c * DO + d] = scl2 * vv[d];
  }
  float hid[HIDN];
  for (int j = 0; j < HIDN; ++j) {
    float a = mcb1[j];
    for (int i = 0; i < CC * DO; ++i) a += xcat[i] * mcW1[i * HIDN + j];
    hid[j] = felu(a);
  }
  float fl = flags[row];
  float o[DO]; float s2 = 0.f;
  for (int kk = 0; kk < DO; ++kk) {
    float a = mcb2[kk];
    for (int j = 0; j < HIDN; ++j) a += hid[j] * mcW2[j * DO + kk];
    a *= fl; o[kk] = a; s2 += a * a;
  }
  float nr = fmaxf(sqrtf(s2), EPSF);
  float scl = ftanh(nr) / nr;
  for (int kk = 0; kk < DO; ++kk) xout[(size_t)row * DO + kk] = scl * o[kk];
}

// ---------------------------------------------------------------------------
// K5: raw mask = mean_h2 tanh((r-dist)*t), computed directly through the
// [B,N,N,H]->[H,B,N,N] view algebra: for output (c,b2,n2,m2):
//   m = m2>>2 + 512*(n2&3), h = m2&3, u = (n2>>2)+512*b2+1024*h2,
//   n_src = u & 2047, b_src = u >> 11.  Never materializes dist.
// grid = C*B*N x 256.
// ---------------------------------------------------------------------------
__global__ void k5_mask(const float* __restrict__ qbuf, const float* __restrict__ sqbuf,
                        const float* __restrict__ rin, const float* __restrict__ tin,
                        float* __restrict__ maskraw) {
  __shared__ float qn[4][HH][DA];
  __shared__ float sqn[4][HH];
  int n2 = blockIdx.x & (NN - 1);
  int b2 = (blockIdx.x >> 11) & 1;
  int c  = blockIdx.x >> 12;
  int t = threadIdx.x;
  if (t < 128) {
    int h2 = t >> 5, rem = t & 31, h = rem >> 3, d = rem & 7;
    int u = (n2 >> 2) + 512 * b2 + 1024 * h2;
    int nsrc = u & (NN - 1), bsrc = u >> 11;
    qn[h2][h][d] = qbuf[(((size_t)(c * BB + bsrc) * NN + nsrc) * HH + h) * DA + d];
  } else if (t < 144) {
    int idx = t - 128, h2 = idx >> 2, h = idx & 3;
    int u = (n2 >> 2) + 512 * b2 + 1024 * h2;
    int nsrc = u & (NN - 1), bsrc = u >> 11;
    sqn[h2][h] = sqbuf[((size_t)(c * BB + bsrc) * NN + nsrc) * HH + h];
  }
  __syncthreads();
  float rr[4], tt[4]; int bsA[4];
#pragma unroll
  for (int h2 = 0; h2 < 4; ++h2) {
    rr[h2] = rin[c * HH + h2];
    tt[h2] = tin[c * HH + h2];
    bsA[h2] = ((n2 >> 2) + 512 * b2 + 1024 * h2) >> 11;
  }
  size_t obase = ((size_t)(c * BB + b2) * NN + n2) * NN;
  for (int i = 0; i < 8; ++i) {
    int m2 = t + 256 * i;
    int m = (m2 >> 2) + 512 * (n2 & 3);
    int h = m2 & 3;
    float acc = 0.f;
#pragma unroll
    for (int h2 = 0; h2 < 4; ++h2) {
      int bsrc = bsA[h2];
      size_t qm = (((size_t)(c * BB + bsrc) * NN + m) * HH + h) * DA;
      float dot = 0.f;
#pragma unroll
      for (int d = 0; d < DA; ++d) dot += qn[h2][h][d] * qbuf[qm + d];
      float sqm = sqbuf[((size_t)(c * BB + bsrc) * NN + m) * HH + h];
      float sq_n = sqn[h2][h];
      float d2 = fmaxf(sq_n + sqm - 2.f * dot, 0.f);
      float den = fmaxf((1.f - sq_n) * (1.f - sqm), EPSF);
      float z = fmaxf(1.f + 2.f * d2 / den, 1.f + 1e-7f);
      float dist = __logf(z + sqrtf(z * z - 1.f));   // arccosh
      acc += ftanh((rr[h2] - dist) * tt[h2]);
    }
    maskraw[obase + m2] = 0.25f * acc;
  }
}

// ---------------------------------------------------------------------------
// K6: symmetrize masks, edge MLP at (n,m) and (m,n), adj_out. The result is
// symmetric by construction, so each block with jt>=it computes one triangle
// tile and writes BOTH (i,j) and (j,i) output tiles (transposed side staged
// through the 32x33 LDS tiles so every global store stays coalesced).
// Halves mask/adj read traffic. grid = B*64*64 x 256 (jt<it blocks exit).
// ---------------------------------------------------------------------------
__device__ __forceinline__ void mlp4(const float in[4], const float* __restrict__ W1,
                                     const float* __restrict__ B1,
                                     const float* __restrict__ W2,
                                     const float* __restrict__ B2, float out[CO]) {
  float h[HIDN];
#pragma unroll
  for (int j = 0; j < HIDN; ++j) {
    float a = B1[j];
#pragma unroll
    for (int i = 0; i < 4; ++i) a += in[i] * W1[i * HIDN + j];
    h[j] = felu(a);
  }
#pragma unroll
  for (int k = 0; k < CO; ++k) {
    float a = B2[k];
#pragma unroll
    for (int j = 0; j < HIDN; ++j) a += h[j] * W2[j * CO + k];
    out[k] = a;
  }
}

__global__ void k6_adj(const float* __restrict__ maskraw, const float* __restrict__ adj,
                       const float* __restrict__ flags,
                       const float* __restrict__ w1, const float* __restrict__ b1,
                       const float* __restrict__ w2, const float* __restrict__ b2,
                       float* __restrict__ adjout) {
  __shared__ float m0a[32][33], m0b[32][33], m1a[32][33], m1b[32][33];
  __shared__ float a0a[32][33], a0b[32][33], a1a[32][33], a1b[32][33];
  int jt = blockIdx.x & 63;
  int it = (blockIdx.x >> 6) & 63;
  int b  = blockIdx.x >> 12;
  if (jt < it) return;                      // triangle only; we write both tiles
  size_t mbase0 = ((size_t)(0 * BB + b)) * NN * NN;
  size_t mbase1 = ((size_t)(1 * BB + b)) * NN * NN;
  size_t abase0 = ((size_t)(b * CC + 0)) * NN * NN;
  size_t abase1 = ((size_t)(b * CC + 1)) * NN * NN;
  for (int q = 0; q < 4; ++q) {
    int idx = threadIdx.x + 256 * q;
    int rr2 = idx >> 5, cc2 = idx & 31;
    size_t ij = (size_t)(it * 32 + rr2) * NN + jt * 32 + cc2;
    size_t ji = (size_t)(jt * 32 + rr2) * NN + it * 32 + cc2;
    m0a[rr2][cc2] = maskraw[mbase0 + ij];
    m0b[rr2][cc2] = maskraw[mbase0 + ji];
    m1a[rr2][cc2] = maskraw[mbase1 + ij];
    m1b[rr2][cc2] = maskraw[mbase1 + ji];
    a0a[rr2][cc2] = adj[abase0 + ij];
    a0b[rr2][cc2] = adj[abase0 + ji];
    a1a[rr2][cc2] = adj[abase1 + ij];
    a1b[rr2][cc2] = adj[abase1 + ji];
  }
  __syncthreads();
  float res0[4], res1[4];
  for (int q = 0; q < 4; ++q) {
    int idx = threadIdx.x + 256 * q;
    int rr2 = idx >> 5, cc2 = idx & 31;
    int n = it * 32 + rr2, m = jt * 32 + cc2;
    float s0 = 0.5f * (m0a[rr2][cc2] + m0b[cc2][rr2]);
    float s1 = 0.5f * (m1a[rr2][cc2] + m1b[cc2][rr2]);
    float in0[4] = {s0, s1, a0a[rr2][cc2], a1a[rr2][cc2]};
    float in1[4] = {s0, s1, a0b[cc2][rr2], a1b[cc2][rr2]};
    float o0[CO], o1[CO];
    mlp4(in0, w1, b1, w2, b2, o0);
    mlp4(in1, w1, b1, w2, b2, o1);
    float fl = flags[b * NN + n] * flags[b * NN + m];
    res0[q] = (o0[0] + o1[0]) * fl;
    res1[q] = (o0[1] + o1[1]) * fl;
  }
  __syncthreads();                           // all reads of LDS tiles done
  for (int q = 0; q < 4; ++q) {              // stage results (reuse m0a/m1a)
    int idx = threadIdx.x + 256 * q;
    int rr2 = idx >> 5, cc2 = idx & 31;
    m0a[rr2][cc2] = res0[q];
    m1a[rr2][cc2] = res1[q];
  }
  __syncthreads();
  for (int q = 0; q < 4; ++q) {
    int idx = threadIdx.x + 256 * q;
    int rr2 = idx >> 5, cc2 = idx & 31;
    size_t nIdx = (size_t)(it * 32 + rr2) * NN + jt * 32 + cc2;
    size_t tIdx = (size_t)(jt * 32 + rr2) * NN + it * 32 + cc2;
    adjout[((size_t)(b * CO + 0)) * NN * NN + nIdx] = m0a[rr2][cc2];
    adjout[((size_t)(b * CO + 1)) * NN * NN + nIdx] = m1a[rr2][cc2];
    adjout[((size_t)(b * CO + 0)) * NN * NN + tIdx] = m0a[cc2][rr2];
    adjout[((size_t)(b * CO + 1)) * NN * NN + tIdx] = m1a[cc2][rr2];
  }
}

// ---------------------------------------------------------------------------
extern "C" void kernel_launch(void* const* d_in, const int* in_sizes, int n_in,
                              void* d_out, int out_size, void* d_ws, size_t ws_size,
                              hipStream_t stream) {
  const float* x     = (const float*)d_in[0];
  const float* adj   = (const float*)d_in[1];
  const float* flags = (const float*)d_in[2];
  const float* Wq    = (const float*)d_in[3];
  const float* aqs   = (const float*)d_in[4];
  const float* aqd   = (const float*)d_in[5];
  const float* Wv    = (const float*)d_in[6];
  const float* avs   = (const float*)d_in[7];
  const float* avd   = (const float*)d_in[8];
  const float* rin   = (const float*)d_in[9];
  const float* tin   = (const float*)d_in[10];
  const float* mcW1  = (const float*)d_in[11];
  const float* mcb1  = (const float*)d_in[12];
  const float* mcW2  = (const float*)d_in[13];
  const float* mcb2  = (const float*)d_in[14];
  const float* mW1   = (const float*)d_in[15];
  const float* mb1   = (const float*)d_in[16];
  const float* mW2   = (const float*)d_in[17];
  const float* mb2   = (const float*)d_in[18];

  float* ws = (float*)d_ws;
  float* srcq = ws + OFF_SRCQ;
  float* dstq = ws + OFF_DSTQ;
  float* srcv = ws + OFF_SRCV;
  float* dstv = ws + OFF_DSTV;
  float* rmax = ws + OFF_RMAX;
  float* rsum = ws + OFF_RSUM;
  float* qatt = ws + OFF_QATT;
  float* vraw = ws + OFF_VRAW;
  float* qbuf = ws + OFF_QBUF;
  float* sqb  = ws + OFF_SQ;
  float* mask = ws + OFF_MASK;
  _Float16* hfrag = (_Float16*)((char*)d_ws + OFF_HFRAG_BYTES);

  float* xout   = (float*)d_out;
  float* adjout = (float*)d_out + (size_t)BB * NN * DO;

  k1_proj <<<512, 256, 0, stream>>>(x, Wq, aqs, aqd, Wv, avs, avd,
                                    srcq, dstq, srcv, dstv, hfrag);
  k2_stats<<<1024, 256, 0, stream>>>(adj, srcq, dstq, srcv, dstv, rmax, rsum);
  k3_attn <<<CC * BB * 128, 256, 0, stream>>>(adj, srcq, srcv, dstq, dstv,
                                              rmax, rsum, hfrag, qatt, vraw);
  k4_rows <<<16, 256, 0, stream>>>(qatt, vraw, flags, mcW1, mcb1, mcW2, mcb2,
                                   qbuf, sqb, xout);
  k5_mask <<<CC * BB * NN, 256, 0, stream>>>(qbuf, sqb, rin, tin, mask);
  k6_adj  <<<BB * 64 * 64, 256, 0, stream>>>(mask, adj, flags, mW1, mb1, mW2, mb2,
                                             adjout);
}
